// MultiHeadAttention_natten_88768384073822
// MI455X (gfx1250) — matrix operand order repaired and checked by reading, not compiled, not run on hardware
//
#include <hip/hip_runtime.h>

// Dilated neighborhood attention (NATTEN), MI455X / gfx1250, wave32 + WMMA f16.
// v2: f16-everywhere data path. One-shot converters give contiguous 128-bit
// fragment loads in every hot loop; V is stored transposed+group-major with a
// zero pad so attention V-fragments are contiguous; GEMM waves do 16x64 tiles
// (A-fragment reuse across 4 WMMAs); global_prefetch on the A stream.

#define B_   2
#define L_   8192
#define E_   256
#define H_   8
#define K_   31
#define DIL_ 2
#define DH_  32
#define LG_  (L_ / DIL_)    // 4096 positions per dilation group
#define LGP_ (LG_ + 64)     // padded row length for transposed V (zero tail)

typedef __attribute__((ext_vector_type(16))) _Float16 v16h;
typedef __attribute__((ext_vector_type(8)))  float    v8f;

// A-fragment (16-bit A, 16x32): lane = m + 16*hi;
// halves e -> K = hi*16 + e (matches B-fragment order).
static __device__ __forceinline__ v16h load_a_f16(const _Float16* rowBase, int hi) {
  v16h a;
  const _Float16* p = rowBase + hi * 16;
#pragma unroll
  for (int e = 0; e < 16; ++e) a[e] = p[e];
  return a;
}
// B-fragment (16-bit B, 32x16): lane = n + 16*hi; halves e -> K = hi*16 + e.
// With N-major f16 storage this is 16 contiguous halves.
static __device__ __forceinline__ v16h load_b_contig(const _Float16* p) {
  v16h b;
#pragma unroll
  for (int e = 0; e < 16; ++e) b[e] = p[e];
  return b;
}

// ---------------------------------------------------------------------------
// One-shot converters (L2-resident, negligible cost).
// ---------------------------------------------------------------------------
__global__ __launch_bounds__(256)
void cvt_f32_to_f16(const float* __restrict__ src, _Float16* __restrict__ dst, int n) {
  const int i = blockIdx.x * 256 + threadIdx.x;
  if (i < n) dst[i] = (_Float16)src[i];
}

// w: [256][N] row-major f32  ->  wT: [N][256] f16 (N-major for B-fragments)
__global__ __launch_bounds__(256)
void cvt_w_transpose(const float* __restrict__ w, _Float16* __restrict__ wT, int N) {
  const int i = blockIdx.x * 256 + threadIdx.x;
  if (i < E_ * N) {
    const int kk = i / N, n = i % N;
    wT[(size_t)n * E_ + kk] = (_Float16)w[i];
  }
}

// ---------------------------------------------------------------------------
// Kernel 1: qkv = x @ w_qkv + b_qkv. M=16384, N=768, K=256.
// One wave computes a 16x64 tile (4 N-tiles share one A fragment).
// Q,K scatter to head-major [b][h][l][dh]; V scatters to transposed
// group-major [b][h][g][dh][p] (padded rows).
// ---------------------------------------------------------------------------
__global__ __launch_bounds__(256)
void qkv_wmma_kernel(const _Float16* __restrict__ xh, const _Float16* __restrict__ wT,
                     const float* __restrict__ bias,
                     _Float16* __restrict__ qb, _Float16* __restrict__ kb,
                     _Float16* __restrict__ vt) {
  const int lane = threadIdx.x & 31;
  const int hi   = lane >> 4;
  const int nlo  = lane & 15;
  const int tile = blockIdx.x * 8 + (threadIdx.x >> 5);   // 12288 waves
  const int NG   = (3 * E_) / 64;                         // 12 n-groups
  const int tm = tile / NG, tg = tile % NG;
  const int m  = tm * 16 + nlo;

  const _Float16* arow = xh + (size_t)m * E_;
  v8f acc[4] = {{}, {}, {}, {}};

#pragma unroll
  for (int k0 = 0; k0 < E_; k0 += 32) {
    if (k0 + 32 < E_) __builtin_prefetch(arow + k0 + 32, 0, 0);
    const v16h a = load_a_f16(arow + k0, hi);
#pragma unroll
    for (int nt = 0; nt < 4; ++nt) {
      const int n = (tg * 4 + nt) * 16 + nlo;
      const v16h bm = load_b_contig(wT + (size_t)n * E_ + k0 + hi * 16);
      acc[nt] = __builtin_amdgcn_wmma_f32_16x16x32_f16(false, a, false, bm,
                                                       (short)0, acc[nt], false, false);
    }
  }

#pragma unroll
  for (int nt = 0; nt < 4; ++nt) {
    const int gn = (tg * 4 + nt) * 16 + nlo;
    const float bv  = bias[gn];
    const int which = gn / E_;                // uniform per n-tile
    const int e2 = gn % E_;
    const int h = e2 / DH_, d = e2 % DH_;
#pragma unroll
    for (int r = 0; r < 8; ++r) {
      const int mm = tm * 16 + r + 8 * hi;    // C/D: M = r + 8*hi
      const int bi = mm / L_, l = mm % L_;
      const _Float16 val = (_Float16)(acc[nt][r] + bv);
      if (which == 2) {                       // V: transposed, group-major, padded
        const int g = l & 1, p = l >> 1;
        vt[((((size_t)bi * H_ + h) * DIL_ + g) * DH_ + d) * LGP_ + p] = val;
      } else {                                // Q/K: head-major rows
        _Float16* dst = which ? kb : qb;
        dst[(((size_t)bi * H_ + h) * L_ + l) * DH_ + d] = val;
      }
    }
  }
}

// ---------------------------------------------------------------------------
// Kernel 2: windowed attention, all-WMMA.
// One wave: (b, h, group g, 16 group positions p0..p0+15); key union = 48 keys
// from s0 = clip(p0-15, 0, Lg-31). S = Q*K^T (3 WMMAs) -> mask/softmax ->
// P via LDS layout flip -> O = P*V (4 WMMAs, contiguous V loads from vt).
// ---------------------------------------------------------------------------
#define PSTR 56

__global__ __launch_bounds__(128)
void natten_attn_kernel(const _Float16* __restrict__ q, const _Float16* __restrict__ k,
                        const _Float16* __restrict__ vt, _Float16* __restrict__ o) {
  __shared__ _Float16 ldsP[4][16 * PSTR];
  const int lane = threadIdx.x & 31;
  const int hi   = lane >> 4;
  const int nlo  = lane & 15;
  const int lw   = threadIdx.x >> 5;
  const int wid  = blockIdx.x * 4 + lw;      // 8192 waves
  const int ptile = wid & 255;
  const int g     = (wid >> 8) & 1;
  const int h     = (wid >> 9) & 7;
  const int b     = wid >> 12;

  const size_t head = ((size_t)b * H_ + h) * L_ * DH_;
  const _Float16* qh = q + head;
  const _Float16* kh = k + head;
  const _Float16* vth = vt + (((size_t)b * H_ + h) * DIL_ + g) * DH_ * LGP_;

  const int p0 = ptile * 16;
  int s0 = p0 - (K_ / 2);
  if (s0 < 0) s0 = 0;
  if (s0 > LG_ - K_) s0 = LG_ - K_;

  // Q fragment
  const int lq = g + DIL_ * (p0 + nlo);
  const v16h qa = load_a_f16(qh + (size_t)lq * DH_, hi);

  // Scores: 3 x (16x16) tiles over 48 union keys (contiguous K rows).
  v8f s[3];
#pragma unroll
  for (int t = 0; t < 3; ++t) {
    int kp = s0 + t * 16 + nlo;
    if (kp > LG_ - 1) kp = LG_ - 1;          // tail masked below
    const v16h kf = load_b_contig(kh + (size_t)(g + DIL_ * kp) * DH_ + hi * 16);
    v8f z = {};
    s[t] = __builtin_amdgcn_wmma_f32_16x16x32_f16(false, qa, false, kf,
                                                  (short)0, z, false, false);
  }

  // Mask each row to its 31-window, scale, softmax.
  const float scale = 0.17677669529663687f;  // 1/sqrt(DH)
  float rmax[8], rsum[8];
#pragma unroll
  for (int r = 0; r < 8; ++r) rmax[r] = -1e30f;
#pragma unroll
  for (int r = 0; r < 8; ++r) {
    const int p = p0 + r + 8 * hi;
    int stp = p - (K_ / 2);
    if (stp < 0) stp = 0;
    if (stp > LG_ - K_) stp = LG_ - K_;
    const int lo = stp - s0;
#pragma unroll
    for (int t = 0; t < 3; ++t) {
      const int j = t * 16 + nlo;
      const bool valid = (j >= lo) && (j < lo + K_);
      const float val = valid ? s[t][r] * scale : -1e30f;
      s[t][r] = val;
      rmax[r] = fmaxf(rmax[r], val);
    }
  }
#pragma unroll
  for (int r = 0; r < 8; ++r)
#pragma unroll
    for (int mw = 1; mw < 16; mw <<= 1)      // stays inside each 16-lane half
      rmax[r] = fmaxf(rmax[r], __shfl_xor(rmax[r], mw, 32));
#pragma unroll
  for (int r = 0; r < 8; ++r) rsum[r] = 0.f;
#pragma unroll
  for (int r = 0; r < 8; ++r)
#pragma unroll
    for (int t = 0; t < 3; ++t) {
      const float pv = __expf(s[t][r] - rmax[r]);
      s[t][r] = pv;
      rsum[r] += pv;
    }
#pragma unroll
  for (int r = 0; r < 8; ++r)
#pragma unroll
    for (int mw = 1; mw < 16; mw <<= 1)
      rsum[r] += __shfl_xor(rsum[r], mw, 32);

  // Normalize; stage P (16x48 f16) in LDS to flip C-layout -> A-layout.
#pragma unroll
  for (int r = 0; r < 8; ++r) {
    const float inv = 1.f / rsum[r];
    const int mrow = r + 8 * hi;
#pragma unroll
    for (int t = 0; t < 3; ++t)
      ldsP[lw][mrow * PSTR + t * 16 + nlo] = (_Float16)(s[t][r] * inv);
  }

  // O = P(16x48 -> pad K 64) x V(48x32): 2 K-chunks x 2 N-tiles.
  // V loads are contiguous thanks to vt layout; pad region is zero.
  v8f oacc[2] = {{}, {}};
#pragma unroll
  for (int kc = 0; kc < 2; ++kc) {
    v16h pa;
    const _Float16* pr = &ldsP[lw][nlo * PSTR];
#pragma unroll
    for (int e = 0; e < 8; ++e) {
      const int j = kc * 32 + hi * 16 + e;
      pa[e] = (j < 48) ? pr[j] : (_Float16)0.f;
    }
#pragma unroll
    for (int e = 0; e < 8; ++e) {
      const int j = kc * 32 + hi * 16 + 8 + e;
      pa[8 + e] = (j < 48) ? pr[j] : (_Float16)0.f;
    }
#pragma unroll
    for (int u = 0; u < 2; ++u) {
      const int d = u * 16 + nlo;
      const v16h vf = load_b_contig(vth + (size_t)d * LGP_ + s0 + kc * 32 + hi * 16);
      oacc[u] = __builtin_amdgcn_wmma_f32_16x16x32_f16(false, pa, false, vf,
                                                       (short)0, oacc[u], false, false);
    }
  }

  // Store O as f16 into [b][l][h*DH + d] (A layout for the proj GEMM).
#pragma unroll
  for (int u = 0; u < 2; ++u) {
    const int d = u * 16 + nlo;
#pragma unroll
    for (int r = 0; r < 8; ++r) {
      const int p = p0 + r + 8 * hi;
      const int l = g + DIL_ * p;
      o[((size_t)b * L_ + l) * E_ + h * DH_ + d] = (_Float16)oacc[u][r];
    }
  }
}

// ---------------------------------------------------------------------------
// Kernel 3: out = attn @ w_proj + b_proj. M=16384, N=256, K=256, f32 out.
// 16x64 tile per wave.
// ---------------------------------------------------------------------------
__global__ __launch_bounds__(256)
void proj_wmma_kernel(const _Float16* __restrict__ attn, const _Float16* __restrict__ wT,
                      const float* __restrict__ bias, float* __restrict__ out) {
  const int lane = threadIdx.x & 31;
  const int hi   = lane >> 4;
  const int nlo  = lane & 15;
  const int tile = blockIdx.x * 8 + (threadIdx.x >> 5);   // 4096 waves
  const int NG   = E_ / 64;                               // 4 n-groups
  const int tm = tile / NG, tg = tile % NG;
  const int m  = tm * 16 + nlo;

  const _Float16* arow = attn + (size_t)m * E_;
  v8f acc[4] = {{}, {}, {}, {}};

#pragma unroll
  for (int k0 = 0; k0 < E_; k0 += 32) {
    if (k0 + 32 < E_) __builtin_prefetch(arow + k0 + 32, 0, 0);
    const v16h a = load_a_f16(arow + k0, hi);
#pragma unroll
    for (int nt = 0; nt < 4; ++nt) {
      const int n = (tg * 4 + nt) * 16 + nlo;
      const v16h bm = load_b_contig(wT + (size_t)n * E_ + k0 + hi * 16);
      acc[nt] = __builtin_amdgcn_wmma_f32_16x16x32_f16(false, a, false, bm,
                                                       (short)0, acc[nt], false, false);
    }
  }

#pragma unroll
  for (int nt = 0; nt < 4; ++nt) {
    const int gn = (tg * 4 + nt) * 16 + nlo;
    const float bv = bias[gn];
#pragma unroll
    for (int r = 0; r < 8; ++r) {
      const int mm = tm * 16 + r + 8 * hi;
      out[(size_t)mm * E_ + gn] = acc[nt][r] + bv;
    }
  }
}

// ---------------------------------------------------------------------------
extern "C" void kernel_launch(void* const* d_in, const int* in_sizes, int n_in,
                              void* d_out, int out_size, void* d_ws, size_t ws_size,
                              hipStream_t stream) {
  const float* x     = (const float*)d_in[0];
  const float* w_qkv = (const float*)d_in[1];
  const float* b_qkv = (const float*)d_in[2];
  const float* w_prj = (const float*)d_in[3];
  const float* b_prj = (const float*)d_in[4];
  float* out = (float*)d_out;

  // Workspace layout (halves):
  const size_t NXE = (size_t)B_ * L_ * E_;                  // 4,194,304
  const size_t NQ  = (size_t)B_ * H_ * L_ * DH_;            // 4,194,304
  const size_t NVT = (size_t)B_ * H_ * DIL_ * DH_ * LGP_;   // 4,259,840
  _Float16* xh     = (_Float16*)d_ws;
  _Float16* qb     = xh + NXE;
  _Float16* kb     = qb + NQ;
  _Float16* vt     = kb + NQ;
  _Float16* at     = vt + NVT;
  _Float16* wqkvT  = at + NXE;
  _Float16* wprjT  = wqkvT + (size_t)E_ * 3 * E_;

  // Zero V (covers the 64-element pad tails; valid region overwritten below).
  hipMemsetAsync(vt, 0, NVT * sizeof(_Float16), stream);

  // One-shot f16 conversion / weight transposition.
  cvt_f32_to_f16<<<(int)(NXE / 256), 256, 0, stream>>>(x, xh, (int)NXE);
  cvt_w_transpose<<<(E_ * 3 * E_) / 256, 256, 0, stream>>>(w_qkv, wqkvT, 3 * E_);
  cvt_w_transpose<<<(E_ * E_) / 256, 256, 0, stream>>>(w_prj, wprjT, E_);

  // 1024 m-tiles * 12 n-groups = 12288 waves / 8 per block
  qkv_wmma_kernel<<<1536, 256, 0, stream>>>(xh, wqkvT, b_qkv, qb, kb, vt);
  // 2*8*2*256 = 8192 waves / 4 per block
  natten_attn_kernel<<<2048, 128, 0, stream>>>(qb, kb, vt, at);
  // 1024 m-tiles * 4 n-groups = 4096 waves / 8 per block
  proj_wmma_kernel<<<512, 256, 0, stream>>>(at, wprjT, b_prj, out);

  (void)in_sizes; (void)n_in; (void)out_size; (void)ws_size;
}
